// GatedGCNLayer_48533130445226
// MI455X (gfx1250) — matrix operand order repaired and checked by reading, not compiled, not run on hardware
//
#include <hip/hip_runtime.h>
#include <hip/hip_bf16.h>

#define NN 50000
#define EE 800000
#define DD 64
#define EPS 1e-5f

typedef __attribute__((ext_vector_type(16))) __bf16 v16bf;
typedef __attribute__((ext_vector_type(8)))  float  v8f;

__device__ __forceinline__ int lane_id() { return (int)(threadIdx.x & 31u); }

// Order LDS round-trips within a wave (store tile -> reload in other layout).
__device__ __forceinline__ void lds_fence() {
  asm volatile("s_wait_dscnt 0" ::: "memory");
  __builtin_amdgcn_wave_barrier();
}

__device__ __forceinline__ v8f wmma_bf16(v16bf a, v16bf b, v8f c) {
  // D = A(16x32 bf16) x B(32x16 bf16) + C(16x16 f32)
  return __builtin_amdgcn_wmma_f32_16x16x32_bf16(false, a, false, b, (short)0, c, false, false);
}

// ---- fragment loaders (CDNA5 wave32 WMMA VGPR layouts, ISA 7.12.2) ----

// A fragment (16x32): this lane owns row m = lane&15; `row` points at that row (f32).
// VGPR p pair holds K = kc*32 + (p<4?0:16) + (lane>>4)*8 + (p&3)*2
__device__ __forceinline__ v16bf load_afrag_f32row(const float* row, int kc) {
  int half = (lane_id() >> 4) & 1;
  v16bf a;
#pragma unroll
  for (int p = 0; p < 8; ++p) {
    int k = kc * 32 + ((p < 4) ? 0 : 16) + half * 8 + (p & 3) * 2;
    a[2 * p]     = (__bf16)row[k];
    a[2 * p + 1] = (__bf16)row[k + 1];
  }
  return a;
}

// A fragment from an LDS bf16 tile (16 rows x 64 cols, row-major)
__device__ __forceinline__ v16bf load_afrag_lds(const __bf16* tile, int kc) {
  int lane = lane_id();
  int half = lane >> 4;
  const __bf16* row = tile + (size_t)(lane & 15) * 64;
  v16bf a;
#pragma unroll
  for (int p = 0; p < 8; ++p) {
    int k = kc * 32 + ((p < 4) ? 0 : 16) + half * 8 + (p & 3) * 2;
    a[2 * p]     = row[k];
    a[2 * p + 1] = row[k + 1];
  }
  return a;
}

// B fragment (32x16) for out = act @ W^T : B[k][n] = W[(ntile*16+n)*ldk + kglob + k]
// lane holds n=lane&15; element pair p holds K = (p<4?0:16) + (lane>>4)*8 + (p&3)*2.
__device__ __forceinline__ v16bf load_bfrag_w(const float* W, int ldk, int ntile, int kglob) {
  int lane = lane_id();
  const float* wrow = W + (size_t)(ntile * 16 + (lane & 15)) * ldk + kglob + (lane >> 4) * 8;
  v16bf b;
#pragma unroll
  for (int p = 0; p < 8; ++p) {
    int k = ((p < 4) ? 0 : 16) + (p & 3) * 2;
    b[2 * p]     = (__bf16)wrow[k];
    b[2 * p + 1] = (__bf16)wrow[k + 1];
  }
  return b;
}

// C/D tile (16x16 f32): lane holds n = lane&15, rows m = (lane>>4)*8 + r
__device__ __forceinline__ void store_ctile_lds_bf16(__bf16* tile, int ntile, v8f c) {
  int lane = lane_id();
  int n  = ntile * 16 + (lane & 15);
  int mb = (lane >> 4) * 8;
#pragma unroll
  for (int r = 0; r < 8; ++r) tile[(size_t)(mb + r) * 64 + n] = (__bf16)c[r];
}

// ---------------------------------------------------------------------------
__global__ void __launch_bounds__(256) k_zero(float* p, int n) {
  int i = blockIdx.x * blockDim.x + threadIdx.x;
  int stride = gridDim.x * blockDim.x;
  for (; i < n; i += stride) p[i] = 0.0f;
}

// Ax,Bx,Dx,Ex = x @ W{A,B,D,E}^T + b ; one 16-row tile per wave, 4 outputs.
__global__ void __launch_bounds__(256) k_node_linear(
    const float* __restrict__ x,
    const float* __restrict__ WA, const float* __restrict__ bA,
    const float* __restrict__ WB, const float* __restrict__ bB,
    const float* __restrict__ WD, const float* __restrict__ bD,
    const float* __restrict__ WE, const float* __restrict__ bE,
    float* __restrict__ ws) {
  const int NT = NN / 16;  // 3125, exact
  int tile = blockIdx.x * 8 + (threadIdx.x >> 5);
  if (tile >= NT) return;  // wave-uniform: EXEC stays all-ones for WMMA
  int lane = lane_id();
  int row0 = tile * 16;
  const float* xrow = x + (size_t)(row0 + (lane & 15)) * DD;
  v16bf a0 = load_afrag_f32row(xrow, 0);
  v16bf a1 = load_afrag_f32row(xrow, 1);

  const float* Ws[4] = {WA, WB, WD, WE};
  const float* bs[4] = {bA, bB, bD, bE};
  int n = lane & 15, mb = (lane >> 4) * 8;
#pragma unroll
  for (int p = 0; p < 4; ++p) {
    float* outp = ws + (size_t)p * NN * DD;
#pragma unroll
    for (int nt = 0; nt < 4; ++nt) {
      v8f c = {};
      c = wmma_bf16(a0, load_bfrag_w(Ws[p], DD, nt, 0), c);
      c = wmma_bf16(a1, load_bfrag_w(Ws[p], DD, nt, 32), c);
      float bias = bs[p][nt * 16 + n];
#pragma unroll
      for (int r = 0; r < 8; ++r)
        outp[(size_t)(row0 + mb + r) * DD + nt * 16 + n] = c[r] + bias;
    }
  }
}

// Fully fused edge pipeline: one 16-edge tile per wave.
__global__ void __launch_bounds__(256) k_edge(
    const int* __restrict__ src, const int* __restrict__ dst,
    const float* __restrict__ edge_attr,
    const float* __restrict__ WC, const float* __restrict__ bC,
    const float* __restrict__ W1, const float* __restrict__ b1,
    const float* __restrict__ W2, const float* __restrict__ b2,
    const float* __restrict__ Bx, const float* __restrict__ Dx,
    const float* __restrict__ Ex,
    float* __restrict__ agg, float* __restrict__ deg,
    float* __restrict__ esum, float* __restrict__ esumsq,
    float* __restrict__ outE) {
  __shared__ __bf16 tileA[8][16 * 64];
  __shared__ int sI[8][16];
  __shared__ int dI[8][16];

  int w = threadIdx.x >> 5;
  int lane = lane_id();
  int half = lane >> 4, n = lane & 15;
  int tile = blockIdx.x * 8 + w;        // 6250 * 8 = 50000 tiles, exact
  int e0 = tile * 16;

  int eOwn = e0 + n;                    // this lane's A-fragment row
  int sOwn = src[eOwn];
  int dOwn = dst[eOwn];
  if (lane < 16) { sI[w][lane] = sOwn; dI[w][lane] = dOwn; }
  lds_fence();

  // gathered A fragments (L2-resident node tensors)
  v16bf aD0 = load_afrag_f32row(Dx + (size_t)dOwn * DD, 0);
  v16bf aD1 = load_afrag_f32row(Dx + (size_t)dOwn * DD, 1);
  v16bf aE0 = load_afrag_f32row(Ex + (size_t)sOwn * DD, 0);
  v16bf aE1 = load_afrag_f32row(Ex + (size_t)sOwn * DD, 1);
  v16bf aT0 = load_afrag_f32row(edge_attr + (size_t)eOwn * DD, 0);
  v16bf aT1 = load_afrag_f32row(edge_attr + (size_t)eOwn * DD, 1);

  // Ce = edge_attr @ WC^T + bC  (kept in f32 regs + bf16 LDS copy)
  v8f ceF[4];
#pragma unroll
  for (int nt = 0; nt < 4; ++nt) {
    v8f c = {};
    c = wmma_bf16(aT0, load_bfrag_w(WC, DD, nt, 0), c);
    c = wmma_bf16(aT1, load_bfrag_w(WC, DD, nt, 32), c);
    float bias = bC[nt * 16 + n];
#pragma unroll
    for (int r = 0; r < 8; ++r) c[r] += bias;
    ceF[nt] = c;
    store_ctile_lds_bf16(tileA[w], nt, c);
  }
  lds_fence();
  v16bf aC0 = load_afrag_lds(tileA[w], 0);
  v16bf aC1 = load_afrag_lds(tileA[w], 1);
  lds_fence();

  // layer 1: relu([Dd | Es | Ce] @ W1^T + b1), K = 192
#pragma unroll
  for (int nt = 0; nt < 4; ++nt) {
    v8f c = {};
    c = wmma_bf16(aD0, load_bfrag_w(W1, 3 * DD, nt, 0), c);
    c = wmma_bf16(aD1, load_bfrag_w(W1, 3 * DD, nt, 32), c);
    c = wmma_bf16(aE0, load_bfrag_w(W1, 3 * DD, nt, 64), c);
    c = wmma_bf16(aE1, load_bfrag_w(W1, 3 * DD, nt, 96), c);
    c = wmma_bf16(aC0, load_bfrag_w(W1, 3 * DD, nt, 128), c);
    c = wmma_bf16(aC1, load_bfrag_w(W1, 3 * DD, nt, 160), c);
    float bias = b1[nt * 16 + n];
#pragma unroll
    for (int r = 0; r < 8; ++r) c[r] = fmaxf(c[r] + bias, 0.0f);
    store_ctile_lds_bf16(tileA[w], nt, c);  // e_mid overwrites Ce tile
  }
  lds_fence();
  v16bf aM0 = load_afrag_lds(tileA[w], 0);
  v16bf aM1 = load_afrag_lds(tileA[w], 1);

  // layer 2: e_pre = e_mid @ W2^T + b2
  v8f ep[4];
#pragma unroll
  for (int nt = 0; nt < 4; ++nt) {
    v8f c = {};
    c = wmma_bf16(aM0, load_bfrag_w(W2, DD, nt, 0), c);
    c = wmma_bf16(aM1, load_bfrag_w(W2, DD, nt, 32), c);
    float bias = b2[nt * 16 + n];
#pragma unroll
    for (int r = 0; r < 8; ++r) c[r] += bias;
    ep[nt] = c;
  }

  // edge-BN stat accumulation: reduce 16 rows of the tile per channel first
#pragma unroll
  for (int nt = 0; nt < 4; ++nt) {
    float ps = 0.0f, pq = 0.0f;
#pragma unroll
    for (int r = 0; r < 8; ++r) { float v = ep[nt][r]; ps += v; pq += v * v; }
    ps += __shfl_xor(ps, 16);
    pq += __shfl_xor(pq, 16);
    if (lane < 16) {
      atomicAdd(&esum[nt * 16 + lane], ps);
      atomicAdd(&esumsq[nt * 16 + lane], pq);
    }
  }

  // write e_pre (normalized later) + gated message aggregation
#pragma unroll
  for (int r = 0; r < 8; ++r) {
    int m2 = half * 8 + r;
    int s2 = sI[w][m2];
    int d2 = dI[w][m2];
    const float* dxr = Dx + (size_t)d2 * DD;
    const float* exr = Ex + (size_t)s2 * DD;
    const float* bxr = Bx + (size_t)s2 * DD;
    float* outr = outE + (size_t)(e0 + m2) * DD;
#pragma unroll
    for (int nt = 0; nt < 4; ++nt) {
      int c = nt * 16 + n;
      outr[c] = ep[nt][r];
      float g = ceF[nt][r] + dxr[c] + exr[c];
      float sig = 1.0f / (1.0f + __expf(-g));
      atomicAdd(agg + (size_t)d2 * DD + c, sig * bxr[c]);
    }
  }
  if (lane < 16) atomicAdd(&deg[dOwn], 1.0f);
}

// h_pre = Ax + agg / max(deg,1); accumulate node-BN stats via LDS pre-reduction
__global__ void __launch_bounds__(256) k_node_hpre(
    const float* __restrict__ Ax, const float* __restrict__ agg,
    const float* __restrict__ deg, float* __restrict__ outH,
    float* __restrict__ nsum, float* __restrict__ nsumsq) {
  __shared__ float ss[64], sq[64];
  int t = threadIdx.x;
  if (t < 64) { ss[t] = 0.0f; sq[t] = 0.0f; }
  __syncthreads();
  const int TOT = NN * DD;
  int base = blockIdx.x * 2048;
#pragma unroll
  for (int k = 0; k < 8; ++k) {
    int i = base + k * 256 + t;
    if (i < TOT) {
      int row = i >> 6, c = i & 63;
      float dg = fmaxf(deg[row], 1.0f);
      float hp = Ax[i] + agg[i] / dg;
      outH[i] = hp;
      atomicAdd(&ss[c], hp);
      atomicAdd(&sq[c], hp * hp);
    }
  }
  __syncthreads();
  if (t < 64) {
    atomicAdd(&nsum[t], ss[t]);
    atomicAdd(&nsumsq[t], sq[t]);
  }
}

// in-place BatchNorm1d (training stats, biased var) + optional relu
__global__ void __launch_bounds__(256) k_norm(
    float* __restrict__ buf, const float* __restrict__ sum,
    const float* __restrict__ sumsq, const float* __restrict__ gamma,
    const float* __restrict__ beta, float count, int total, int do_relu) {
  int i = blockIdx.x * blockDim.x + threadIdx.x;
  if (i >= total) return;
  int c = i & 63;
  float mu = sum[c] / count;
  float var = sumsq[c] / count - mu * mu;
  float rs = rsqrtf(var + EPS);
  float v = gamma[c] * (buf[i] - mu) * rs + beta[c];
  if (do_relu) v = fmaxf(v, 0.0f);
  buf[i] = v;
}

extern "C" void kernel_launch(void* const* d_in, const int* in_sizes, int n_in,
                              void* d_out, int out_size, void* d_ws, size_t ws_size,
                              hipStream_t stream) {
  const float* x     = (const float*)d_in[0];
  const int*   ei    = (const int*)d_in[1];      // [0..E)=src, [E..2E)=dst
  const float* eattr = (const float*)d_in[2];
  const float* WA = (const float*)d_in[3];  const float* bA = (const float*)d_in[4];
  const float* WB = (const float*)d_in[5];  const float* bB = (const float*)d_in[6];
  const float* WD = (const float*)d_in[7];  const float* bD = (const float*)d_in[8];
  const float* WE = (const float*)d_in[9];  const float* bE = (const float*)d_in[10];
  const float* WC = (const float*)d_in[11]; const float* bC = (const float*)d_in[12];
  const float* W1 = (const float*)d_in[13]; const float* b1 = (const float*)d_in[14];
  const float* W2 = (const float*)d_in[15]; const float* b2 = (const float*)d_in[16];
  const float* gn = (const float*)d_in[17]; const float* bn = (const float*)d_in[18];
  const float* ge = (const float*)d_in[19]; const float* be = (const float*)d_in[20];

  float* out = (float*)d_out;                 // [0..N*D) h, [N*D..N*D+E*D) e_new
  float* ws = (float*)d_ws;
  const size_t ND = (size_t)NN * DD;
  float* Ax  = ws;
  float* Bx  = ws + 1 * ND;
  float* Dxp = ws + 2 * ND;
  float* Exp = ws + 3 * ND;
  float* agg = ws + 4 * ND;
  float* deg = ws + 5 * ND;
  float* stats = deg + NN;                    // nsum|nsumsq|esum|esumsq, 64 each

  // zero agg + deg + stats (contiguous)
  k_zero<<<512, 256, 0, stream>>>(agg, (int)(ND + NN + 256));

  // node linears (WMMA bf16)
  k_node_linear<<<391, 256, 0, stream>>>(x, WA, bA, WB, bB, WD, bD, WE, bE, ws);

  // fused edge pipeline (WMMA bf16): Ce, gate, MLP, BN stats, messages
  k_edge<<<6250, 256, 0, stream>>>(ei, ei + EE, eattr, WC, bC, W1, b1, W2, b2,
                                   Bx, Dxp, Exp, agg, deg,
                                   stats + 128, stats + 192, out + ND);

  // h_pre + node BN stats
  k_node_hpre<<<1563, 256, 0, stream>>>(Ax, agg, deg, out, stats, stats + 64);

  // normalize h (with relu) and e_new in place
  k_norm<<<(int)((ND + 255) / 256), 256, 0, stream>>>(
      out, stats, stats + 64, gn, bn, (float)NN, (int)ND, 1);
  k_norm<<<(int)(((size_t)EE * DD + 255) / 256), 256, 0, stream>>>(
      out + ND, stats + 128, stats + 192, ge, be, (float)EE, EE * DD, 0);
}
